// rnn_three_20048907337990
// MI455X (gfx1250) — matrix operand order repaired and checked by reading, not compiled, not run on hardware
//
#include <hip/hip_runtime.h>

// MI455X / gfx1250 fused implementation.
//   Kernel 1: per-sequence LSTM-1 (hidden 7, T=10) with the per-step affine
//             z = [x_t, h, 1] @ [W1;U1;b1] done on v_wmma_f32_16x16x32_f16
//             (f16 in / f32 accumulate), + Wd1 dot + coef + dc-mean via shfl.
//             Streams the 335 MB input with non-temporal b128 loads.
//   Kernel 2: per-sequence LSTM-2 (hidden 10, T=10), same WMMA machinery
//             (3 N-tiles for the 40 gate columns), final hidden -> Wd2 dot.
// Workspace: 2048*128*10 floats (10.5 MB) for the stage-1 -> stage-2 tensor.

typedef __attribute__((ext_vector_type(16))) _Float16     v16h;
typedef __attribute__((ext_vector_type(8)))  float        v8f;
typedef __attribute__((ext_vector_type(4)))  float        v4f;
typedef __attribute__((ext_vector_type(4)))  unsigned int v4u;

#define N_DIMS 128
#define BATCH  2048
#define LL     10
#define DC     4
#define RW     8

union Pack16 {
  _Float16 h[16];
  v4u      q[2];
  v16h     v;
};

__device__ __forceinline__ float fast_tanh(float x) {
#if __has_builtin(__builtin_amdgcn_tanhf)
  return __builtin_amdgcn_tanhf(x);                 // native v_tanh_f32 on gfx1250
#elif __has_builtin(__builtin_amdgcn_exp2f) && __has_builtin(__builtin_amdgcn_rcpf)
  float e = __builtin_amdgcn_exp2f(-2.885390082f * x);   // exp(-2x)
  return (1.0f - e) * __builtin_amdgcn_rcpf(1.0f + e);
#else
  float e = __expf(-2.0f * x);
  return (1.0f - e) / (1.0f + e);
#endif
}
__device__ __forceinline__ float fast_sig(float x) {
  return 0.5f * fast_tanh(0.5f * x) + 0.5f;
}

// ---------------------------------------------------------------- kernel 1 --
__global__ __launch_bounds__(256) void lstm1_kernel(
    const float* __restrict__ trim, const float* __restrict__ W1,
    const float* __restrict__ U1,   const float* __restrict__ b1,
    const float* __restrict__ Wd1,  const float* __restrict__ bd1,
    const float* __restrict__ coef, float* __restrict__ x2out) {
  __shared__ __align__(16) _Float16 xhS[256][16];   // per-sample [x(7), h(7), 1, 0] f16
  __shared__ float zS[8][32][36];                   // per-wave [gate][sample] f32 (padded)

  const int tid  = threadIdx.x;
  const int lane = tid & 31;
  const int wv   = tid >> 5;
  const int s    = blockIdx.x * 256 + tid;          // sample = ((n*B + b)*4 + d)

  // Constant B fragments: combined weights Wc[16][32], Wc[k][c]:
  //   k<7 -> W1[k][c], k<14 -> U1[k-7][c], k==14 -> b1[c], k==15 -> 0; c>=28 -> 0.
  // B layout used here (contiguous k): lanes 0-15 hold K=0..15 for col=lane; lanes 16-31 hold
  // K=16..31 (all zero padding here).
  v16h bf0 = {}, bf1 = {};
  if (lane < 16) {
    const int c0 = lane, c1 = lane + 16;
#pragma unroll
    for (int k = 0; k < 15; ++k) {
      float w0, w1;
      if (k < 7)       { w0 = W1[k * 28 + c0];       w1 = (c1 < 28) ? W1[k * 28 + c1]       : 0.f; }
      else if (k < 14) { w0 = U1[(k - 7) * 28 + c0]; w1 = (c1 < 28) ? U1[(k - 7) * 28 + c1] : 0.f; }
      else             { w0 = b1[c0];                w1 = (c1 < 28) ? b1[c1]                : 0.f; }
      bf0[k] = (_Float16)w0;
      bf1[k] = (_Float16)w1;
    }
  }

  float wdv[7];
#pragma unroll
  for (int j = 0; j < 7; ++j) wdv[j] = Wd1[j];
  const float bdv = bd1[0];

  float c[7], h[7];
#pragma unroll
  for (int j = 0; j < 7; ++j) { c[j] = 0.f; h[j] = 0.f; }

  // trim[n, b*L + t, d, r] -> float offset (s>>2)*320 + t*32 + (s&3)*8
  const long rowBase = (long)(s >> 2) * (LL * DC * RW) + (long)(s & 3) * RW;

  for (int t = 0; t < LL; ++t) {
    v4f r0 = __builtin_nontemporal_load((const v4f*)(trim + rowBase + t * 32));
    v4f r1 = __builtin_nontemporal_load((const v4f*)(trim + rowBase + t * 32 + 4));
    const float y = r0[0];

    Pack16 pk;
    pk.h[0] = (_Float16)r0[1]; pk.h[1] = (_Float16)r0[2]; pk.h[2] = (_Float16)r0[3];
    pk.h[3] = (_Float16)r1[0]; pk.h[4] = (_Float16)r1[1]; pk.h[5] = (_Float16)r1[2];
    pk.h[6] = (_Float16)r1[3];
#pragma unroll
    for (int j = 0; j < 7; ++j) pk.h[7 + j] = (_Float16)h[j];
    pk.h[14] = (_Float16)1.0f;
    pk.h[15] = (_Float16)0.0f;
    *(v4u*)&xhS[tid][0] = pk.q[0];
    *(v4u*)&xhS[tid][8] = pk.q[1];
    __syncthreads();

    // A fragments (contiguous layout: lanes 0-15 K0..15; lanes 16-31 multiply zero B)
    Pack16 a0, a1;
    const int half = (lane >> 4) * 8;
    const v4u zq = {0u, 0u, 0u, 0u};
    a0.q[0] = *(const v4u*)&xhS[wv * 32 +      (lane & 15)][0];
    a1.q[0] = *(const v4u*)&xhS[wv * 32 + 16 + (lane & 15)][0];
    a0.q[1] = *(const v4u*)&xhS[wv * 32 +      (lane & 15)][8];
    a1.q[1] = *(const v4u*)&xhS[wv * 32 + 16 + (lane & 15)][8];

    v8f acc00 = {}, acc01 = {}, acc10 = {}, acc11 = {};
    acc00 = __builtin_amdgcn_wmma_f32_16x16x32_f16(false, a0.v, false, bf0, (short)0, acc00, false, false);
    acc01 = __builtin_amdgcn_wmma_f32_16x16x32_f16(false, a0.v, false, bf1, (short)0, acc01, false, false);
    acc10 = __builtin_amdgcn_wmma_f32_16x16x32_f16(false, a1.v, false, bf0, (short)0, acc10, false, false);
    acc11 = __builtin_amdgcn_wmma_f32_16x16x32_f16(false, a1.v, false, bf1, (short)0, acc11, false, false);

    // D layout: lane = gate col (mod 16), VGPR j = sample row j + 8*(lane>=16).
    // Scatter to zS[wave][gate][sampleLocal]; rows of 8 samples are contiguous.
    const int grow = (lane & 15);
    const int colb = (lane >> 4) * 8;
    *(v4f*)&zS[wv][grow     ][colb     ] = __builtin_shufflevector(acc00, acc00, 0, 1, 2, 3);
    *(v4f*)&zS[wv][grow     ][colb + 4 ] = __builtin_shufflevector(acc00, acc00, 4, 5, 6, 7);
    *(v4f*)&zS[wv][grow + 16][colb     ] = __builtin_shufflevector(acc01, acc01, 0, 1, 2, 3);
    *(v4f*)&zS[wv][grow + 16][colb + 4 ] = __builtin_shufflevector(acc01, acc01, 4, 5, 6, 7);
    *(v4f*)&zS[wv][grow     ][colb + 16] = __builtin_shufflevector(acc10, acc10, 0, 1, 2, 3);
    *(v4f*)&zS[wv][grow     ][colb + 20] = __builtin_shufflevector(acc10, acc10, 4, 5, 6, 7);
    *(v4f*)&zS[wv][grow + 16][colb + 16] = __builtin_shufflevector(acc11, acc11, 0, 1, 2, 3);
    *(v4f*)&zS[wv][grow + 16][colb + 20] = __builtin_shufflevector(acc11, acc11, 4, 5, 6, 7);
    __syncthreads();

    // Gate math (Keras order i,f,g,o) + Wd1 dot, all per-lane.
    float v = bdv;
#pragma unroll
    for (int j = 0; j < 7; ++j) {
      const float zi = zS[wv][     j][lane];
      const float zf = zS[wv][ 7 + j][lane];
      const float zg = zS[wv][14 + j][lane];
      const float zo = zS[wv][21 + j][lane];
      const float cc = fast_sig(zf) * c[j] + fast_sig(zi) * fast_tanh(zg);
      c[j] = cc;
      const float hh = fast_sig(zo) * fast_tanh(cc);
      h[j] = hh;
      v += hh * wdv[j];
    }

    float oe = y + coef[t] * v;
    oe += __shfl_xor(oe, 1, 32);      // reduce over dc (4 consecutive lanes)
    oe += __shfl_xor(oe, 2, 32);
    if ((s & 3) == 0) {
      const int sb = s >> 2;          // n*B + b
      const int nd = sb >> 11;
      const int b  = sb & (BATCH - 1);
      x2out[(b * N_DIMS + nd) * LL + t] = oe * 0.25f;   // mean over dc, [B, n, L]
    }
  }
}

// ---------------------------------------------------------------- kernel 2 --
__device__ __forceinline__ float wc2val(int k, int ccol, const float* W2,
                                        const float* U2, const float* b2) {
  if (k == 0)  return W2[ccol];
  if (k < 11)  return U2[(k - 1) * 40 + ccol];
  return b2[ccol];
}

__global__ __launch_bounds__(256) void lstm2_kernel(
    const float* __restrict__ x2, const float* __restrict__ W2,
    const float* __restrict__ U2, const float* __restrict__ b2,
    const float* __restrict__ Wd2, const float* __restrict__ bd2,
    float* __restrict__ out) {
  __shared__ __align__(16) _Float16 xhS[256][16];   // [x_t, h(10), 1, 0...]
  __shared__ float zS[8][48][36];                   // 40 gate cols padded to 48

  const int tid  = threadIdx.x;
  const int lane = tid & 31;
  const int wv   = tid >> 5;
  const int s2   = blockIdx.x * 256 + tid;          // sample = b*n_dims + n

  // Combined weights Wc2[16][48]: k==0 -> W2 row, k in [1,11) -> U2[k-1], k==11 -> b2.
  v16h bf0 = {}, bf1 = {}, bf2 = {};
  if (lane < 16) {
#pragma unroll
    for (int k = 0; k < 12; ++k) {
      bf0[k] = (_Float16)wc2val(k, lane,      W2, U2, b2);
      bf1[k] = (_Float16)wc2val(k, lane + 16, W2, U2, b2);
      bf2[k] = (_Float16)((lane + 32 < 40) ? wc2val(k, lane + 32, W2, U2, b2) : 0.f);
    }
  }

  float wdv[10];
#pragma unroll
  for (int j = 0; j < 10; ++j) wdv[j] = Wd2[j];
  const float bdv = bd2[0];

  float c[10], h[10];
#pragma unroll
  for (int j = 0; j < 10; ++j) { c[j] = 0.f; h[j] = 0.f; }

  for (int t = 0; t < LL; ++t) {
    const float xt = x2[(long)s2 * LL + t];

    Pack16 pk;
    pk.h[0] = (_Float16)xt;
#pragma unroll
    for (int j = 0; j < 10; ++j) pk.h[1 + j] = (_Float16)h[j];
    pk.h[11] = (_Float16)1.0f;
    pk.h[12] = (_Float16)0.0f; pk.h[13] = (_Float16)0.0f;
    pk.h[14] = (_Float16)0.0f; pk.h[15] = (_Float16)0.0f;
    *(v4u*)&xhS[tid][0] = pk.q[0];
    *(v4u*)&xhS[tid][8] = pk.q[1];
    __syncthreads();

    Pack16 a0, a1;
    const int half = (lane >> 4) * 8;
    const v4u zq = {0u, 0u, 0u, 0u};
    a0.q[0] = *(const v4u*)&xhS[wv * 32 +      (lane & 15)][0];
    a1.q[0] = *(const v4u*)&xhS[wv * 32 + 16 + (lane & 15)][0];
    a0.q[1] = *(const v4u*)&xhS[wv * 32 +      (lane & 15)][8];
    a1.q[1] = *(const v4u*)&xhS[wv * 32 + 16 + (lane & 15)][8];

    v8f a00 = {}, a01 = {}, a02 = {}, a10 = {}, a11 = {}, a12 = {};
    a00 = __builtin_amdgcn_wmma_f32_16x16x32_f16(false, a0.v, false, bf0, (short)0, a00, false, false);
    a01 = __builtin_amdgcn_wmma_f32_16x16x32_f16(false, a0.v, false, bf1, (short)0, a01, false, false);
    a02 = __builtin_amdgcn_wmma_f32_16x16x32_f16(false, a0.v, false, bf2, (short)0, a02, false, false);
    a10 = __builtin_amdgcn_wmma_f32_16x16x32_f16(false, a1.v, false, bf0, (short)0, a10, false, false);
    a11 = __builtin_amdgcn_wmma_f32_16x16x32_f16(false, a1.v, false, bf1, (short)0, a11, false, false);
    a12 = __builtin_amdgcn_wmma_f32_16x16x32_f16(false, a1.v, false, bf2, (short)0, a12, false, false);

    const int grow = (lane & 15);
    const int colb = (lane >> 4) * 8;
    *(v4f*)&zS[wv][grow     ][colb     ] = __builtin_shufflevector(a00, a00, 0, 1, 2, 3);
    *(v4f*)&zS[wv][grow     ][colb + 4 ] = __builtin_shufflevector(a00, a00, 4, 5, 6, 7);
    *(v4f*)&zS[wv][grow + 16][colb     ] = __builtin_shufflevector(a01, a01, 0, 1, 2, 3);
    *(v4f*)&zS[wv][grow + 16][colb + 4 ] = __builtin_shufflevector(a01, a01, 4, 5, 6, 7);
    *(v4f*)&zS[wv][grow + 32][colb     ] = __builtin_shufflevector(a02, a02, 0, 1, 2, 3);
    *(v4f*)&zS[wv][grow + 32][colb + 4 ] = __builtin_shufflevector(a02, a02, 4, 5, 6, 7);
    *(v4f*)&zS[wv][grow     ][colb + 16] = __builtin_shufflevector(a10, a10, 0, 1, 2, 3);
    *(v4f*)&zS[wv][grow     ][colb + 20] = __builtin_shufflevector(a10, a10, 4, 5, 6, 7);
    *(v4f*)&zS[wv][grow + 16][colb + 16] = __builtin_shufflevector(a11, a11, 0, 1, 2, 3);
    *(v4f*)&zS[wv][grow + 16][colb + 20] = __builtin_shufflevector(a11, a11, 4, 5, 6, 7);
    *(v4f*)&zS[wv][grow + 32][colb + 16] = __builtin_shufflevector(a12, a12, 0, 1, 2, 3);
    *(v4f*)&zS[wv][grow + 32][colb + 20] = __builtin_shufflevector(a12, a12, 4, 5, 6, 7);
    __syncthreads();

#pragma unroll
    for (int j = 0; j < 10; ++j) {
      const float zi = zS[wv][     j][lane];
      const float zf = zS[wv][10 + j][lane];
      const float zg = zS[wv][20 + j][lane];
      const float zo = zS[wv][30 + j][lane];
      const float cc = fast_sig(zf) * c[j] + fast_sig(zi) * fast_tanh(zg);
      c[j] = cc;
      h[j] = fast_sig(zo) * fast_tanh(cc);
    }
  }

  float o = bdv;
#pragma unroll
  for (int j = 0; j < 10; ++j) o += h[j] * wdv[j];
  out[s2] = o;
}

// ------------------------------------------------------------------ launch --
extern "C" void kernel_launch(void* const* d_in, const int* in_sizes, int n_in,
                              void* d_out, int out_size, void* d_ws, size_t ws_size,
                              hipStream_t stream) {
  (void)in_sizes; (void)n_in; (void)out_size; (void)ws_size;

  const float* trim = (const float*)d_in[0];
  const float* W1   = (const float*)d_in[1];
  const float* U1   = (const float*)d_in[2];
  const float* b1   = (const float*)d_in[3];
  const float* Wd1  = (const float*)d_in[4];
  const float* bd1  = (const float*)d_in[5];
  const float* coef = (const float*)d_in[6];
  const float* W2   = (const float*)d_in[7];
  const float* U2   = (const float*)d_in[8];
  const float* b2   = (const float*)d_in[9];
  const float* Wd2  = (const float*)d_in[10];
  const float* bd2  = (const float*)d_in[11];

  float* x2buf = (float*)d_ws;            // B * n_dims * L floats = 10.5 MB
  float* out   = (float*)d_out;           // [B, n_dims] fp32

  const int N1 = N_DIMS * BATCH * DC;     // 1,048,576 sequences
  const int N2 = BATCH * N_DIMS;          //   262,144 sequences

  lstm1_kernel<<<N1 / 256, 256, 0, stream>>>(trim, W1, U1, b1, Wd1, bd1, coef, x2buf);
  lstm2_kernel<<<N2 / 256, 256, 0, stream>>>(x2buf, W2, U2, b2, Wd2, bd2, out);
}
